// ToMeBlock_14302241095816
// MI455X (gfx1250) — matrix operand order repaired and checked by reading, not compiled, not run on hardware
//
#include <hip/hip_runtime.h>
#include <hip/hip_bf16.h>

// ------------------------------------------------------------------
// CLIP-style transformer block + ToMe merge for gfx1250 (MI455X).
// GEMMs via V_WMMA_F32_16X16X32_BF16 (wave32 layouts, ISA 7.12.2).
// Double-buffered GLOBAL_LOAD_ASYNC_TO_LDS_B128 pipeline (ASYNCcnt)
// overlaps the global->LDS DMA of tile k+1 with WMMA on tile k.
// bf16 inputs, fp32 accumulation.
// ------------------------------------------------------------------

typedef __attribute__((ext_vector_type(16))) __bf16          v16bf;
typedef __attribute__((ext_vector_type(16))) unsigned short  v16us;
typedef __attribute__((ext_vector_type(8)))  float           v8f;

static constexpr int BB  = 32;     // batch
static constexpr int T   = 577;    // tokens
static constexpr int C   = 1024;   // channels
static constexpr int H   = 16;     // heads
static constexpr int Dh  = 64;     // head dim
static constexpr int DFF = 4096;
static constexpr int R   = 18;     // merged pairs
static constexpr int NT  = BB * T;           // 18464 rows
static constexpr int SRC = 289;              // even tokens
static constexpr int DST = 288;              // odd tokens
static constexpr int UNM = SRC - R;          // 271
static constexpr int TM  = UNM + DST;        // 559 merged tokens
static constexpr int NTM = BB * TM;          // 17888 rows
static constexpr int TP  = 608;              // padded key count (19*32)

#define NEG_INF (-__builtin_inff())

__device__ __forceinline__ unsigned short f2bf(float f) {
  return (unsigned short)(__float_as_uint(f) >> 16);
}
__device__ __forceinline__ float bf2f(unsigned short h) {
  return __uint_as_float(((unsigned)h) << 16);
}
__device__ __forceinline__ v16bf as_bf(v16us u) {
  return __builtin_bit_cast(v16bf, u);
}

// async 16-byte global -> LDS copy (ASYNCcnt-tracked, EXEC all-ones here)
__device__ __forceinline__ void async_g2l_b128(void* lds, const void* gaddr) {
  unsigned l = (unsigned)(size_t)lds;   // low 32 bits = LDS byte offset
  asm volatile("global_load_async_to_lds_b128 %0, %1, off"
               :: "v"(l), "v"(gaddr) : "memory");
}
// wait until <=4 async transfers outstanding (current tile done, next in flight)
__device__ __forceinline__ void wait_async_4() {
  asm volatile("s_wait_asynccnt 4" ::: "memory");
}
__device__ __forceinline__ void wait_async_0() {
  asm volatile("s_wait_asynccnt 0" ::: "memory");
}

// ---------------- fp32 -> bf16 conversion (weights) ----------------
__global__ void f32_to_bf16(const float* __restrict__ x,
                            unsigned short* __restrict__ y, size_t n) {
  size_t i = (size_t)blockIdx.x * blockDim.x + threadIdx.x;
  size_t stride = (size_t)gridDim.x * blockDim.x;
  for (; i < n; i += stride) y[i] = f2bf(x[i]);
}

// ---------------- LayerNorm (one 256-thread block per row) ---------
__global__ void __launch_bounds__(256)
ln_bf16(const float* __restrict__ x, const float* __restrict__ g,
        const float* __restrict__ bta, unsigned short* __restrict__ out) {
  const int row = blockIdx.x;
  const float* xr = x + (size_t)row * C;
  __shared__ float red[256];
  float s = 0.f;
  for (int c = threadIdx.x; c < C; c += 256) s += xr[c];
  red[threadIdx.x] = s; __syncthreads();
  for (int st = 128; st > 0; st >>= 1) {
    if (threadIdx.x < st) red[threadIdx.x] += red[threadIdx.x + st];
    __syncthreads();
  }
  const float mu = red[0] * (1.f / C);
  __syncthreads();
  float v = 0.f;
  for (int c = threadIdx.x; c < C; c += 256) { float d = xr[c] - mu; v += d * d; }
  red[threadIdx.x] = v; __syncthreads();
  for (int st = 128; st > 0; st >>= 1) {
    if (threadIdx.x < st) red[threadIdx.x] += red[threadIdx.x + st];
    __syncthreads();
  }
  const float rs = rsqrtf(red[0] * (1.f / C) + 1e-5f);
  for (int c = threadIdx.x; c < C; c += 256)
    out[(size_t)row * C + c] = f2bf((xr[c] - mu) * rs * g[c] + bta[c]);
}

// ---------------- Generic batched bf16 WMMA GEMM -------------------
// Out[z][M,N] = epilogue( (A[z][M,K] @ W[z][K,N] + bias) * alpha )
// EPI: 0 = none, 1 = quickGELU, 2 = += resid
// outBT (optional): transposed bf16 out as [b][h][d][t] with t-pitch TP
// Block tile 128x128, 8 waves (4Mx2N), wave tile 32x64, K-step 32.
// Double-buffered async DMA pipeline: tile k+1 DMAs while tile k computes.
template <int EPI>
__global__ void __launch_bounds__(256)
gemm_bf16(const unsigned short* __restrict__ A,
          const unsigned short* __restrict__ W,
          const float* __restrict__ bias,
          const float* __restrict__ resid,
          float* __restrict__ outF, unsigned short* __restrict__ outB,
          unsigned short* __restrict__ outBT,
          int M, int N, int K, float alpha,
          long long strideA, long long strideW, long long strideO) {
  const int bM = blockIdx.x * 128, bN = blockIdx.y * 128;
  const int z = blockIdx.z;
  A += (size_t)z * strideA;
  W += (size_t)z * strideW;
  const size_t obase = (size_t)z * strideO;

  __shared__ unsigned short As[2][128][32];     // 2 x 8 KB
  __shared__ unsigned short Bs[2][32][136];     // 2 x 8.5 KB (+16B row pad)

  const int tid = threadIdx.x;
  const int w = tid >> 5, lane = tid & 31;
  const int wm = w & 3, wn = w >> 2;            // 4x2 waves -> 32x64 per wave
  const int half = lane >> 4, l16 = lane & 15;

  v8f acc[2][4] = {};

  // issue one K-tile's async DMA (4 wave-instructions: 2 A + 2 W chunks/thread)
  auto issue_tile = [&](int it, int buf) {
    const int k0 = it * 32;
#pragma unroll
    for (int i = 0; i < 2; i++) {
      int cid = tid + i * 256;                  // 512 chunks = 128 rows x 4
      int r = cid >> 2, c8 = (cid & 3) * 8;
      int gr = bM + r; if (gr >= M) gr = M - 1; // clamp; results discarded
      async_g2l_b128(&As[buf][r][c8], A + (size_t)gr * K + k0 + c8);
    }
#pragma unroll
    for (int i = 0; i < 2; i++) {
      int cid = tid + i * 256;                  // 512 chunks = 32 rows x 16
      int r = cid >> 4, c8 = (cid & 15) * 8;
      int gc = bN + c8; if (gc >= N) gc = 0;    // clamp; cols >= N discarded
      async_g2l_b128(&Bs[buf][r][c8], W + (size_t)(k0 + r) * N + gc);
    }
  };

  const int nK = K / 32;                         // K % 32 == 0 for all GEMMs
  issue_tile(0, 0);

  for (int it = 0; it < nK; ++it) {
    const int cur = it & 1;
    if (it + 1 < nK) {
      issue_tile(it + 1, cur ^ 1);              // overlap DMA with compute
      if (it + 2 < nK)                          // warm L2 two tiles ahead
        __builtin_prefetch(W + (size_t)((it + 2) * 32 + (tid >> 4)) * N + bN +
                               (tid & 15) * 8, 0, 1);
      wait_async_4();                           // current tile complete
    } else {
      wait_async_0();
    }
    __syncthreads();

    // B fragments (lane-half = K 0-15 / 16-31, lane&15 = column)
    v16us bu[4];
#pragma unroll
    for (int nt = 0; nt < 4; nt++) {
      int col = wn * 64 + nt * 16 + l16;
#pragma unroll
      for (int i = 0; i < 16; i++) bu[nt][i] = Bs[cur][half * 16 + i][col];
    }
#pragma unroll
    for (int mt = 0; mt < 2; mt++) {
      // A fragment: 2 contiguous 16B chunks -> ds_load_b128
      v16us au;
      int r = wm * 32 + mt * 16 + l16;
#pragma unroll
      for (int i = 0; i < 8; i++) {
        au[i]     = As[cur][r][half * 16 + i];
        au[8 + i] = As[cur][r][half * 16 + 8 + i];
      }
      v16bf a = as_bf(au);
#pragma unroll
      for (int nt = 0; nt < 4; nt++) {
        acc[mt][nt] = __builtin_amdgcn_wmma_f32_16x16x32_bf16(
            false, a, false, as_bf(bu[nt]), (short)0, acc[mt][nt], false, false);
      }
    }
    __syncthreads();                            // all reads done before reuse
  }

  // epilogue (C/D layout: lane-half selects row group of 8, col = lane&15)
#pragma unroll
  for (int mt = 0; mt < 2; mt++) {
#pragma unroll
    for (int nt = 0; nt < 4; nt++) {
      int col = bN + wn * 64 + nt * 16 + l16;
#pragma unroll
      for (int e = 0; e < 8; e++) {
        int row = bM + wm * 32 + mt * 16 + half * 8 + e;
        if (row < M && col < N) {
          float v = acc[mt][nt][e];
          if (bias) v += bias[col];
          v *= alpha;
          if (EPI == 1) v = v / (1.f + __expf(-1.702f * v));  // quickGELU
          size_t o = obase + (size_t)row * N + col;
          if (EPI == 2 && resid) v += resid[o];
          if (outF) outF[o] = v;
          if (outB) outB[o] = f2bf(v);
          if (outBT) {                           // [b][h][d][t], t-pitch TP
            int b = row / T, t = row - b * T;
            int h = col >> 6, d = col & 63;
            outBT[(((size_t)b * H + h) * Dh + d) * TP + t] = f2bf(v);
          }
        }
      }
    }
  }
}

// ---------------- Attention: one wave per (16 queries, head, batch) -----
// q,k: [B][T][C] bf16 ; vt: [B][H][Dh][TP] bf16 (key-contiguous)
__global__ void __launch_bounds__(32)
attention(const unsigned short* __restrict__ q,
          const unsigned short* __restrict__ k,
          const unsigned short* __restrict__ vt,
          unsigned short* __restrict__ o) {
  __shared__ float S[16][TP];             // 38.9 KB of 320 KB/WGP

  const int qb = blockIdx.x, h = blockIdx.y, b = blockIdx.z;
  const int lane = threadIdx.x;
  const int half = lane >> 4, l16 = lane & 15;
  const size_t base = (size_t)b * T * C + (size_t)h * Dh;
  const int q0 = qb * 16;

  // init padded score rows to -inf (exp -> 0)
  for (int r = 0; r < 16; r++)
    for (int c = lane; c < TP; c += 32) S[r][c] = NEG_INF;

  // Q block -> two A-register sets (K-dim = 64 = 2 x 32)
  int qr = q0 + l16; if (qr >= T) qr = T - 1;
  const unsigned short* qrow = q + base + (size_t)qr * C;
  v16us qa0, qa1;
#pragma unroll
  for (int i = 0; i < 8; i++) {
    qa0[i]     = qrow[half * 16 + i];
    qa0[8 + i] = qrow[half * 16 + 8 + i];
    qa1[i]     = qrow[32 + half * 16 + i];
    qa1[8 + i] = qrow[32 + half * 16 + 8 + i];
  }
  const v16bf qA0 = as_bf(qa0), qA1 = as_bf(qa1);

  // pass 1: S = (q*scale) @ k^T  (scale folded into q already)
  for (int kb = 0; kb < T; kb += 16) {
    int key = kb + l16;
    int kc = key < T ? key : T - 1;
    const unsigned short* krow = k + base + (size_t)kc * C;
    v16us b0, b1;
#pragma unroll
    for (int i = 0; i < 16; i++) {
      b0[i] = krow[half * 16 + i];        // B[d][key], d = half*16+i
      b1[i] = krow[32 + half * 16 + i];
    }
    v8f s = {};
    s = __builtin_amdgcn_wmma_f32_16x16x32_bf16(false, qA0, false, as_bf(b0),
                                                (short)0, s, false, false);
    s = __builtin_amdgcn_wmma_f32_16x16x32_bf16(false, qA1, false, as_bf(b1),
                                                (short)0, s, false, false);
    if (key < T) {
#pragma unroll
      for (int e = 0; e < 8; e++) S[half * 8 + e][key] = s[e];
    }
  }
  __syncthreads();

  // softmax: 2 lanes per row (pair (l16, l16+16)), combine via shfl_xor 16
  {
    const int row = l16;
    float m = NEG_INF;
    for (int c = half; c < T; c += 2) m = fmaxf(m, S[row][c]);
    m = fmaxf(m, __shfl_xor(m, 16, 32));
    float sum = 0.f;
    for (int c = half; c < T; c += 2) {
      float e = __expf(S[row][c] - m);
      S[row][c] = e; sum += e;
    }
    sum += __shfl_xor(sum, 16, 32);
    const float inv = 1.f / sum;
    for (int c = half; c < TP; c += 2) S[row][c] = (c < T) ? S[row][c] * inv : 0.f;
  }
  __syncthreads();

  // pass 2: O = P @ V  (keys in chunks of 32; padded keys have p == 0)
  const unsigned short* vhead = vt + ((size_t)b * H + h) * Dh * TP;
  v8f oacc[4] = {};
  for (int kb = 0; kb < TP; kb += 32) {
    v16us pu;
    const int row = l16;
#pragma unroll
    for (int i = 0; i < 8; i++) {
      pu[i]     = f2bf(S[row][kb + half * 16 + i]);
      pu[8 + i] = f2bf(S[row][kb + half * 16 + 8 + i]);
    }
    const v16bf pA = as_bf(pu);
#pragma unroll
    for (int nt = 0; nt < 4; nt++) {
      // lane = output dim d, element i = key -> contiguous 32B runs
      const unsigned short* vrow = vhead + (size_t)(nt * 16 + l16) * TP + kb + half * 16;
      v16us vu;
#pragma unroll
      for (int i = 0; i < 16; i++) vu[i] = vrow[i];
      oacc[nt] = __builtin_amdgcn_wmma_f32_16x16x32_bf16(
          false, pA, false, as_bf(vu), (short)0, oacc[nt], false, false);
    }
  }
#pragma unroll
  for (int nt = 0; nt < 4; nt++) {
#pragma unroll
    for (int e = 0; e < 8; e++) {
      int row = q0 + half * 8 + e;
      if (row < T) o[base + (size_t)row * C + nt * 16 + l16] = f2bf(oacc[nt][e]);
    }
  }
}

// ---------------- ToMe: normalize metric, split into a / b^T -------
__global__ void __launch_bounds__(256)
metric_norm(const unsigned short* __restrict__ kmat,
            unsigned short* __restrict__ aBuf,     // [B][SRC][C]
            unsigned short* __restrict__ bTBuf) {  // [B][C][DST]
  const int t = blockIdx.x % T, b = blockIdx.x / T;
  const unsigned short* row = kmat + ((size_t)b * T + t) * C;
  __shared__ float red[256];
  float s = 0.f;
  for (int c = threadIdx.x; c < C; c += 256) { float x = bf2f(row[c]); s += x * x; }
  red[threadIdx.x] = s; __syncthreads();
  for (int st = 128; st > 0; st >>= 1) {
    if (threadIdx.x < st) red[threadIdx.x] += red[threadIdx.x + st];
    __syncthreads();
  }
  const float inv = rsqrtf(fmaxf(red[0], 1e-30f));
  if (t & 1) {
    const int j = t >> 1;
    for (int c = threadIdx.x; c < C; c += 256)
      bTBuf[((size_t)b * C + c) * DST + j] = f2bf(bf2f(row[c]) * inv);
  } else {
    const int u = t >> 1;
    for (int c = threadIdx.x; c < C; c += 256)
      aBuf[((size_t)b * SRC + u) * C + c] = f2bf(bf2f(row[c]) * inv);
  }
}

// ---------------- ToMe: argmax + bitonic argsort + index sets ------
__global__ void __launch_bounds__(256)
tome_select(const float* __restrict__ scores,   // [B][SRC][DST]
            int* __restrict__ unm, int* __restrict__ srcI,
            int* __restrict__ dstI) {
  const int b = blockIdx.x;
  const float* sc = scores + (size_t)b * SRC * DST;
  __shared__ float nmax[SRC];
  __shared__ int   nidx[SRC];
  __shared__ float key[512];
  __shared__ int   kid[512];
  __shared__ int   un[512];

  for (int r = threadIdx.x; r < SRC; r += 256) {
    float m = NEG_INF; int mi = 0;
    if (r != 0) {                       // scores[:,0,:] forced to -inf
      for (int c = 0; c < DST; c++) {
        float x = sc[(size_t)r * DST + c];
        if (x > m) { m = x; mi = c; }
      }
    }
    nmax[r] = m; nidx[r] = mi;
  }
  __syncthreads();

  for (int i = threadIdx.x; i < 512; i += 256) {
    if (i < SRC) { key[i] = nmax[i]; kid[i] = i; }
    else         { key[i] = NEG_INF; kid[i] = 0x7FFFFFFF; }
  }
  __syncthreads();

  // bitonic sort: descending by key, ascending by id on ties
  for (int ksz = 2; ksz <= 512; ksz <<= 1) {
    for (int j = ksz >> 1; j > 0; j >>= 1) {
      for (int i = threadIdx.x; i < 512; i += 256) {
        int ixj = i ^ j;
        if (ixj > i) {
          bool down = ((i & ksz) == 0);
          float a = key[i], c = key[ixj];
          int   ai = kid[i], ci = kid[ixj];
          bool aFirst = (a > c) || (a == c && ai < ci);
          bool swp = down ? !aFirst : aFirst;
          if (swp) { key[i] = c; key[ixj] = a; kid[i] = ci; kid[ixj] = ai; }
        }
      }
      __syncthreads();
    }
  }

  if (threadIdx.x < R) {
    int s = kid[threadIdx.x];
    srcI[b * R + threadIdx.x] = s;
    dstI[b * R + threadIdx.x] = nidx[s];
  }
  __syncthreads();

  for (int i = threadIdx.x; i < 512; i += 256)
    un[i] = (i < UNM) ? kid[R + i] : 0x7FFFFFFF;
  __syncthreads();

  // bitonic ascending on unmerged indices
  for (int ksz = 2; ksz <= 512; ksz <<= 1) {
    for (int j = ksz >> 1; j > 0; j >>= 1) {
      for (int i = threadIdx.x; i < 512; i += 256) {
        int ixj = i ^ j;
        if (ixj > i) {
          bool up = ((i & ksz) == 0);
          int a = un[i], c = un[ixj];
          bool swp = up ? (a > c) : (a < c);
          if (swp) { un[i] = c; un[ixj] = a; }
        }
      }
      __syncthreads();
    }
  }
  for (int i = threadIdx.x; i < UNM; i += 256) unm[b * UNM + i] = un[i];
}

// ---------------- Merge: gather / scatter-mean ---------------------
__global__ void __launch_bounds__(256)
merge_gather(const float* __restrict__ hidden, const int* __restrict__ unm,
             float* __restrict__ merged, float* __restrict__ cnt) {
  const int row = blockIdx.x, b = blockIdx.y;
  int tok = (row < UNM) ? 2 * unm[b * UNM + row] : 2 * (row - UNM) + 1;
  const float* s = hidden + ((size_t)b * T + tok) * C;
  float* d = merged + ((size_t)b * TM + row) * C;
  for (int c = threadIdx.x; c < C; c += 256) d[c] = s[c];
  if (row >= UNM && threadIdx.x == 0) cnt[b * DST + (row - UNM)] = 1.f;
}

__global__ void __launch_bounds__(256)
merge_scatter(const float* __restrict__ hidden, const int* __restrict__ srcI,
              const int* __restrict__ dstI, float* __restrict__ merged,
              float* __restrict__ cnt) {
  const int i = blockIdx.x, b = blockIdx.y;
  const int s = srcI[b * R + i], d = dstI[b * R + i];
  const float* src = hidden + ((size_t)b * T + 2 * s) * C;
  float* dst = merged + ((size_t)b * TM + UNM + d) * C;
  for (int c = threadIdx.x; c < C; c += 256) atomicAdd(&dst[c], src[c]);
  if (threadIdx.x == 0) atomicAdd(&cnt[b * DST + d], 1.f);
}

__global__ void __launch_bounds__(256)
merge_div(float* __restrict__ merged, const float* __restrict__ cnt) {
  const int j = blockIdx.x, b = blockIdx.y;
  const float inv = 1.f / cnt[b * DST + j];
  float* d = merged + ((size_t)b * TM + UNM + j) * C;
  for (int c = threadIdx.x; c < C; c += 256) d[c] *= inv;
}

// ------------------------------------------------------------------
extern "C" void kernel_launch(void* const* d_in, const int* in_sizes, int n_in,
                              void* d_out, int out_size, void* d_ws, size_t ws_size,
                              hipStream_t stream) {
  const float* hs    = (const float*)d_in[0];
  const float* ln1_g = (const float*)d_in[1];
  const float* ln1_b = (const float*)d_in[2];
  const float* q_w   = (const float*)d_in[3];
  const float* q_b   = (const float*)d_in[4];
  const float* k_w   = (const float*)d_in[5];
  const float* k_b   = (const float*)d_in[6];
  const float* v_w   = (const float*)d_in[7];
  const float* v_b   = (const float*)d_in[8];
  const float* o_w   = (const float*)d_in[9];
  const float* o_b   = (const float*)d_in[10];
  const float* ln2_g = (const float*)d_in[11];
  const float* ln2_b = (const float*)d_in[12];
  const float* fc1_w = (const float*)d_in[13];
  const float* fc1_b = (const float*)d_in[14];
  const float* fc2_w = (const float*)d_in[15];
  const float* fc2_b = (const float*)d_in[16];

  char* ws = (char*)d_ws;
  size_t off = 0;
  auto alloc = [&](size_t bytes) -> void* {
    off = (off + 255) & ~(size_t)255;
    void* p = ws + off;
    off += bytes;
    return p;
  };

  unsigned short* Wq   = (unsigned short*)alloc((size_t)C * C * 2);
  unsigned short* Wk   = (unsigned short*)alloc((size_t)C * C * 2);
  unsigned short* Wv   = (unsigned short*)alloc((size_t)C * C * 2);
  unsigned short* Wo   = (unsigned short*)alloc((size_t)C * C * 2);
  unsigned short* Wf1  = (unsigned short*)alloc((size_t)C * DFF * 2);
  unsigned short* Wf2  = (unsigned short*)alloc((size_t)DFF * C * 2);
  unsigned short* xn   = (unsigned short*)alloc((size_t)NT * C * 2);
  unsigned short* qB   = (unsigned short*)alloc((size_t)NT * C * 2);
  unsigned short* kB   = (unsigned short*)alloc((size_t)NT * C * 2);
  unsigned short* vT   = (unsigned short*)alloc((size_t)BB * H * Dh * TP * 2);
  unsigned short* oB   = (unsigned short*)alloc((size_t)NT * C * 2);
  float*          hid  = (float*)alloc((size_t)NT * C * 4);
  unsigned short* aBuf = (unsigned short*)alloc((size_t)BB * SRC * C * 2);
  unsigned short* bT   = (unsigned short*)alloc((size_t)BB * C * DST * 2);
  float*          scr  = (float*)alloc((size_t)BB * SRC * DST * 4);
  int*            unmI = (int*)alloc((size_t)BB * UNM * 4);
  int*            srcI = (int*)alloc((size_t)BB * R * 4);
  int*            dstI = (int*)alloc((size_t)BB * R * 4);
  float*          cnt  = (float*)alloc((size_t)BB * DST * 4);
  float*          mrg  = (float*)alloc((size_t)NTM * C * 4);
  unsigned short* ln2o = (unsigned short*)alloc((size_t)NTM * C * 2);
  unsigned short* f1o  = (unsigned short*)alloc((size_t)NTM * DFF * 2);
  (void)ws_size; (void)in_sizes; (void)n_in; (void)out_size;

  // weights -> bf16
  f32_to_bf16<<<1024, 256, 0, stream>>>(q_w,   Wq,  (size_t)C * C);
  f32_to_bf16<<<1024, 256, 0, stream>>>(k_w,   Wk,  (size_t)C * C);
  f32_to_bf16<<<1024, 256, 0, stream>>>(v_w,   Wv,  (size_t)C * C);
  f32_to_bf16<<<1024, 256, 0, stream>>>(o_w,   Wo,  (size_t)C * C);
  f32_to_bf16<<<1024, 256, 0, stream>>>(fc1_w, Wf1, (size_t)C * DFF);
  f32_to_bf16<<<1024, 256, 0, stream>>>(fc2_w, Wf2, (size_t)DFF * C);

  // LN1
  ln_bf16<<<NT, 256, 0, stream>>>(hs, ln1_g, ln1_b, xn);

  // QKV projections (scale 1/sqrt(D) folded into q; v written transposed)
  dim3 gQKV((NT + 127) / 128, C / 128, 1);
  gemm_bf16<0><<<gQKV, 256, 0, stream>>>(xn, Wq, q_b, nullptr, nullptr, qB,
                                         nullptr, NT, C, C, 0.125f, 0, 0, 0);
  gemm_bf16<0><<<gQKV, 256, 0, stream>>>(xn, Wk, k_b, nullptr, nullptr, kB,
                                         nullptr, NT, C, C, 1.0f, 0, 0, 0);
  gemm_bf16<0><<<gQKV, 256, 0, stream>>>(xn, Wv, v_b, nullptr, nullptr, nullptr,
                                         vT, NT, C, C, 1.0f, 0, 0, 0);

  // attention
  dim3 gAtt((T + 15) / 16, H, BB);
  attention<<<gAtt, 32, 0, stream>>>(qB, kB, vT, oB);

  // output projection + residual
  gemm_bf16<2><<<gQKV, 256, 0, stream>>>(oB, Wo, o_b, hs, hid, nullptr,
                                         nullptr, NT, C, C, 1.0f, 0, 0, 0);

  // ToMe: metric = k; normalize & split even/odd (odd transposed)
  metric_norm<<<NT, 256, 0, stream>>>(kB, aBuf, bT);

  // batched scores GEMM: [SRC,C] @ [C,DST] per batch
  dim3 gScore((SRC + 127) / 128, (DST + 127) / 128, BB);
  gemm_bf16<0><<<gScore, 256, 0, stream>>>(
      aBuf, bT, nullptr, nullptr, scr, nullptr, nullptr, SRC, DST, C, 1.0f,
      (long long)SRC * C, (long long)C * DST, (long long)SRC * DST);

  // selection + merge
  tome_select<<<BB, 256, 0, stream>>>(scr, unmI, srcI, dstI);
  dim3 gGat(TM, BB);  merge_gather<<<gGat, 256, 0, stream>>>(hid, unmI, mrg, cnt);
  dim3 gSct(R, BB);   merge_scatter<<<gSct, 256, 0, stream>>>(hid, srcI, dstI, mrg, cnt);
  dim3 gDiv(DST, BB); merge_div<<<gDiv, 256, 0, stream>>>(mrg, cnt);

  // MLP
  ln_bf16<<<NTM, 256, 0, stream>>>(mrg, ln2_g, ln2_b, ln2o);
  dim3 gF1((NTM + 127) / 128, DFF / 128, 1);
  gemm_bf16<1><<<gF1, 256, 0, stream>>>(ln2o, Wf1, fc1_b, nullptr, nullptr, f1o,
                                        nullptr, NTM, DFF, C, 1.0f, 0, 0, 0);
  dim3 gF2((NTM + 127) / 128, C / 128, 1);
  gemm_bf16<2><<<gF2, 256, 0, stream>>>(f1o, Wf2, fc2_b, mrg, (float*)d_out,
                                        nullptr, nullptr, NTM, C, DFF, 1.0f, 0, 0, 0);
}
